// LSTM_40346922778627
// MI455X (gfx1250) — compile-verified, matrix operand order checked by reading, not run on hardware
//
#include <hip/hip_runtime.h>

#define B_  64
#define T_  512
#define I_  512
#define H_  1024
#define G4  4096   // 4*H
#define KC  1536   // I + H

typedef __attribute__((ext_vector_type(16))) __bf16        v16bf;
typedef __attribute__((ext_vector_type(8)))  float         v8f;
typedef __attribute__((ext_vector_type(4)))  unsigned int  v4u;

union Frag { v16bf v; v4u q[2]; };

struct KSet { Frag a[4]; Frag b0, b1; };

__device__ __forceinline__ unsigned short f2bf(float f) {
  unsigned int u = __float_as_uint(f);
  u += 0x7FFFu + ((u >> 16) & 1u);          // round-to-nearest-even
  return (unsigned short)(u >> 16);
}
__device__ __forceinline__ float sigm(float x) {
  return 1.0f / (1.0f + __expf(-x));
}
__device__ __forceinline__ float tanh_fast(float x) {
  float e = __expf(2.0f * x);
  return (e - 1.0f) / (e + 1.0f);
}

// ---------------- prep kernels ----------------

// Wt[n][k] = W[k][n], n in [0,4096) over gates [i,f,c~,o], k in [0,1536) = [U rows | V rows]
__global__ void k_build_wt(const float* __restrict__ U0, const float* __restrict__ V0,
                           const float* __restrict__ U1, const float* __restrict__ V1,
                           const float* __restrict__ U2, const float* __restrict__ V2,
                           const float* __restrict__ U3, const float* __restrict__ V3,
                           unsigned short* __restrict__ Wt) {
  size_t idx = (size_t)blockIdx.x * blockDim.x + threadIdx.x;
  if (idx >= (size_t)G4 * KC) return;
  int k = (int)(idx % KC);
  int n = (int)(idx / KC);
  int g = n >> 10, nl = n & (H_ - 1);
  const float* U = (g == 0) ? U0 : (g == 1) ? U1 : (g == 2) ? U2 : U3;
  const float* V = (g == 0) ? V0 : (g == 1) ? V1 : (g == 2) ? V2 : V3;
  float f = (k < I_) ? U[(size_t)k * H_ + nl] : V[(size_t)(k - I_) * H_ + nl];
  Wt[idx] = f2bf(f);
}

__global__ void k_build_bias(const float* __restrict__ b0, const float* __restrict__ b1,
                             const float* __restrict__ b2, const float* __restrict__ b3,
                             float* __restrict__ biasC) {
  int n = blockIdx.x * blockDim.x + threadIdx.x;
  if (n >= G4) return;
  int g = n >> 10, nl = n & (H_ - 1);
  biasC[n] = (g == 0) ? b0[nl] : (g == 1) ? b1[nl] : (g == 2) ? b2[nl] : b3[nl];
}

__global__ void k_cvt_x(const float* __restrict__ x, unsigned short* __restrict__ xbf, size_t n) {
  size_t idx = (size_t)blockIdx.x * blockDim.x + threadIdx.x;
  if (idx < n) xbf[idx] = f2bf(x[idx]);
}

__global__ void k_init(unsigned short* __restrict__ hA, unsigned short* __restrict__ hB,
                       float* __restrict__ outp) {
  int idx = blockIdx.x * blockDim.x + threadIdx.x;
  if (idx < 2 * B_ * H_) outp[idx] = 0.0f;
  if (idx < B_ * H_) { hA[idx] = 0; hB[idx] = 0; }
}

// ---------------- recurrent step ----------------

__device__ __forceinline__ void load_set(KSet& s,
                                         const unsigned short* const* xrow,
                                         const unsigned short* const* hrow,
                                         const unsigned short* brow0,
                                         const unsigned short* brow1,
                                         int ks, int kh8) {
  int off = ks * 32 + kh8;
  // B rows are contiguous across the whole K = [U | V] range
  s.b0.q[0] = *(const v4u*)(brow0 + off);
  s.b0.q[1] = *(const v4u*)(brow0 + off + 16);
  s.b1.q[0] = *(const v4u*)(brow1 + off);
  s.b1.q[1] = *(const v4u*)(brow1 + off + 16);
  #pragma unroll
  for (int mt = 0; mt < 4; ++mt) {
    const unsigned short* ar = (ks < 16) ? (xrow[mt] + off) : (hrow[mt] + (off - I_));
    s.a[mt].q[0] = *(const v4u*)(ar);
    s.a[mt].q[1] = *(const v4u*)(ar + 16);
  }
}

__device__ __forceinline__ void mma_set(const KSet& s, v8f acc[4][2]) {
  #pragma unroll
  for (int mt = 0; mt < 4; ++mt) {
    acc[mt][0] = __builtin_amdgcn_wmma_f32_16x16x32_bf16(
        false, s.a[mt].v, false, s.b0.v, (short)0, acc[mt][0], false, false);
    acc[mt][1] = __builtin_amdgcn_wmma_f32_16x16x32_bf16(
        false, s.a[mt].v, false, s.b1.v, (short)0, acc[mt][1], false, false);
  }
}

// grid = 16 WGs (64 H-columns each), block = 256 threads = 8 waves.
// wave w: gate = w>>1, column-half nh = w&1 (32 cols), all 4 m-tiles (64 batch rows).
__global__ __launch_bounds__(256, 1) void k_lstm_step(
    const unsigned short* __restrict__ xbf,   // [B, T, I]  bf16
    const unsigned short* __restrict__ Wt,    // [4096, 1536] bf16 (transposed weights)
    const float* __restrict__ biasC,          // [4096]
    const unsigned short* __restrict__ hprev, // [B, H] bf16
    unsigned short* __restrict__ hnext,       // [B, H] bf16
    float* __restrict__ hout,                 // [B, H] f32  (d_out)
    float* __restrict__ cstate,               // [B, H] f32  (d_out + B*H)
    int t)
{
  __shared__ float gbuf[4 * 64 * 64];         // 64 KB: [gate][row][col]

  const int tid   = threadIdx.x;
  const int wave  = tid >> 5;
  const int lane  = tid & 31;
  const int kh8   = (lane >> 4) * 8;          // K-half offset (A/B frag layout)
  const int l16   = lane & 15;
  const int gate  = wave >> 1;
  const int nh    = wave & 1;
  const int wgcol = blockIdx.x << 6;          // 64-col block within H

  const int ncol0 = gate * H_ + wgcol + nh * 32 + l16;   // n-tile 0
  const int ncol1 = ncol0 + 16;                          // n-tile 1
  const unsigned short* brow0 = Wt + (size_t)ncol0 * KC;
  const unsigned short* brow1 = Wt + (size_t)ncol1 * KC;

  const unsigned short* xrow[4];
  const unsigned short* hrow[4];
  #pragma unroll
  for (int mt = 0; mt < 4; ++mt) {
    int m = mt * 16 + l16;                    // batch row for this lane
    xrow[mt] = xbf + ((size_t)m * T_ + (size_t)t) * I_;
    hrow[mt] = hprev + (size_t)m * H_;
  }

  v8f acc[4][2] = {};

  // ---- K loop over 48 k-steps of 32, 2-deep software pipeline ----
  KSet s0, s1;
  load_set(s0, xrow, hrow, brow0, brow1, 0, kh8);
  load_set(s1, xrow, hrow, brow0, brow1, 1, kh8);
  for (int ks = 0; ks < 46; ks += 2) {
    mma_set(s0, acc);
    load_set(s0, xrow, hrow, brow0, brow1, ks + 2, kh8);
    mma_set(s1, acc);
    load_set(s1, xrow, hrow, brow0, brow1, ks + 3, kh8);
  }
  mma_set(s0, acc);   // ks = 46
  mma_set(s1, acc);   // ks = 47

  // ---- bias + stage pre-activations to LDS ----
  const float bv0 = biasC[ncol0];
  const float bv1 = biasC[ncol1];
  const int colL = nh * 32 + l16;             // 0..63 within the WG's column block
  #pragma unroll
  for (int mt = 0; mt < 4; ++mt) {
    #pragma unroll
    for (int e = 0; e < 8; ++e) {
      int row = mt * 16 + e + kh8;            // C/D layout: M = e + 8*(lane/16)
      gbuf[(gate * 64 + row) * 64 + colL]      = acc[mt][0][e] + bv0;
      gbuf[(gate * 64 + row) * 64 + colL + 16] = acc[mt][1][e] + bv1;
    }
  }
  __syncthreads();

  // ---- fused gates: c_new = sig(i)*tanh(c~) + sig(f)*c ; h = sig(o)*tanh(c_new) ----
  #pragma unroll
  for (int i = 0; i < 16; ++i) {
    int idx = i * 256 + tid;                  // 0..4095 over [64 rows][64 cols]
    int r  = idx >> 6;
    int cL = idx & 63;
    float gi = gbuf[(0 * 64 + r) * 64 + cL];
    float gf = gbuf[(1 * 64 + r) * 64 + cL];
    float gc = gbuf[(2 * 64 + r) * 64 + cL];
    float go = gbuf[(3 * 64 + r) * 64 + cL];
    size_t o = (size_t)r * H_ + wgcol + cL;
    float cold = cstate[o];
    float cn = sigm(gi) * tanh_fast(gc) + sigm(gf) * cold;
    float hn = sigm(go) * tanh_fast(cn);
    cstate[o] = cn;
    hout[o]   = hn;
    hnext[o]  = f2bf(hn);
  }
}

// ---------------- host launcher ----------------
extern "C" void kernel_launch(void* const* d_in, const int* in_sizes, int n_in,
                              void* d_out, int out_size, void* d_ws, size_t ws_size,
                              hipStream_t stream) {
  const float* x  = (const float*)d_in[0];
  const float* U0 = (const float*)d_in[1];
  const float* V0 = (const float*)d_in[2];
  const float* b0 = (const float*)d_in[3];
  const float* U1 = (const float*)d_in[4];
  const float* V1 = (const float*)d_in[5];
  const float* b1 = (const float*)d_in[6];
  const float* U2 = (const float*)d_in[7];
  const float* V2 = (const float*)d_in[8];
  const float* b2 = (const float*)d_in[9];
  const float* U3 = (const float*)d_in[10];
  const float* V3 = (const float*)d_in[11];
  const float* b3 = (const float*)d_in[12];

  char* ws = (char*)d_ws;
  size_t off = 0;
  unsigned short* Wt    = (unsigned short*)(ws + off); off += (size_t)G4 * KC * 2;       // 12.6 MB
  float*          biasC = (float*)(ws + off);          off += (size_t)G4 * 4;            // 16 KB
  unsigned short* xbf   = (unsigned short*)(ws + off); off += (size_t)B_ * T_ * I_ * 2;  // 33.6 MB
  unsigned short* hA    = (unsigned short*)(ws + off); off += (size_t)B_ * H_ * 2;       // 128 KB
  unsigned short* hB    = (unsigned short*)(ws + off); off += (size_t)B_ * H_ * 2;       // 128 KB

  float* hout = (float*)d_out;                       // first B*H floats: h_T
  float* cst  = (float*)d_out + (size_t)B_ * H_;     // next  B*H floats: c_T (live state)

  size_t nwt = (size_t)G4 * KC;
  k_build_wt<<<dim3((unsigned)((nwt + 255) / 256)), dim3(256), 0, stream>>>(
      U0, V0, U1, V1, U2, V2, U3, V3, Wt);
  k_build_bias<<<dim3((G4 + 255) / 256), dim3(256), 0, stream>>>(b0, b1, b2, b3, biasC);
  size_t nx = (size_t)B_ * T_ * I_;
  k_cvt_x<<<dim3((unsigned)((nx + 255) / 256)), dim3(256), 0, stream>>>(x, xbf, nx);
  k_init<<<dim3((2 * B_ * H_ + 255) / 256), dim3(256), 0, stream>>>(hA, hB, (float*)d_out);

  for (int t = 0; t < T_; ++t) {
    const unsigned short* hp = (t & 1) ? hB : hA;
    unsigned short*       hn = (t & 1) ? hA : hB;
    k_lstm_step<<<dim3(H_ / 64), dim3(256), 0, stream>>>(
        xbf, Wt, biasC, hp, hn, hout, cst, t);
  }
}
